// uniform_loss_31619549233204
// MI455X (gfx1250) — matrix operand order repaired and checked by reading, not compiled, not run on hardware
//
#include <hip/hip_runtime.h>
#include <hip/hip_bf16.h>

typedef __attribute__((ext_vector_type(16))) _Float16 v16h;
typedef __attribute__((ext_vector_type(8)))  _Float16 v8h;
typedef __attribute__((ext_vector_type(8)))  float    v8f;

#define UL_N 16384
#define UL_D 128
#define UL_CSPLIT 4                      // column-sweep slices per row-block
#define UL_CCOLS  (UL_N / UL_CSPLIT)     // 4096 columns per slice

// exp(s/T) = exp2(s * 1/(T*ln2)); v_exp_f32 is natively exp2.
#define UL_EXP2_SCALE ((float)(1.0 / (0.07 * 0.69314718055994530942)))
#define UL_LN2 0.69314718055994530942f

// ---------------------------------------------------------------------------
// Kernel 1: split f32 x into f16 hi + f16 residual lo (split-precision GEMM:
// hi*hi + hi*lo + lo*hi gives ~2^-22 relative accuracy on dot products).
// ---------------------------------------------------------------------------
__global__ void ul_split_kernel(const float* __restrict__ x,
                                _Float16* __restrict__ xhi,
                                _Float16* __restrict__ xlo) {
    int idx = blockIdx.x * blockDim.x + threadIdx.x;
    if (idx < UL_N * UL_D) {
        float v = x[idx];
        _Float16 hi = (_Float16)v;
        _Float16 lo = (_Float16)(v - (float)hi);
        xhi[idx] = hi;
        xlo[idx] = lo;
    }
}

// Load a 16x32 f16 A-fragment chunk for this lane.
// Lane layout (contiguous K order): lane m (m=lane&15),
// kh=lane>>4: elems 0..15 = K (16*kh .. 16*kh+15).
__device__ __forceinline__ v16h ul_load_afrag(const _Float16* __restrict__ rowp,
                                              int koff) {
    v8h a0 = *(const v8h*)(rowp + koff);
    v8h a1 = *(const v8h*)(rowp + koff + 8);
    return __builtin_shufflevector(a0, a1,
                                   0, 1, 2, 3, 4, 5, 6, 7,
                                   8, 9, 10, 11, 12, 13, 14, 15);
}

// ---------------------------------------------------------------------------
// Kernel 2: fused X·X^T (WMMA f16 split-precision) + exp partial row-sums.
// Grid: (N/128 row-blocks) x (UL_CSPLIT column slices). 256 threads = 8
// waves; each wave owns a 16-row strip and sweeps its slice's 4096 columns
// in 16-wide tiles; K=128 as 4 chunks of 32, 3 WMMAs per chunk.
// ---------------------------------------------------------------------------
__global__ void __launch_bounds__(256)
ul_main_kernel(const _Float16* __restrict__ xhi,
               const _Float16* __restrict__ xlo,
               float* __restrict__ rowpart) {
    const int lane = threadIdx.x & 31;
    const int wave = threadIdx.x >> 5;
    const int m    = lane & 15;
    const int kh   = lane >> 4;

    const int rowbase = (blockIdx.x * 8 + wave) * 16;
    const int jbase   = blockIdx.y * UL_CCOLS;

    // Preload all A fragments for this wave's 16 rows (resident in VGPRs).
    const _Float16* arow_hi = xhi + (size_t)(rowbase + m) * UL_D;
    const _Float16* arow_lo = xlo + (size_t)(rowbase + m) * UL_D;
    v16h Ahi[4], Alo[4];
#pragma unroll
    for (int c = 0; c < 4; ++c) {
        Ahi[c] = ul_load_afrag(arow_hi, 32 * c + 16 * kh);
        Alo[c] = ul_load_afrag(arow_lo, 32 * c + 16 * kh);
    }

    // Per-lane partial exp-sums: acc[v] covers C-matrix row (v + 8*kh),
    // column (lane&15) of every tile.
    float acc[8];
#pragma unroll
    for (int v = 0; v < 8; ++v) acc[v] = 0.0f;

    // B-fragment per-lane offset: column n = lane&15 -> row x[j+n], 16
    // contiguous f16 at K-offset 16*kh within each 32-wide chunk.
    const size_t bstride = (size_t)m * UL_D + 16 * kh;

    for (int j = jbase; j < jbase + UL_CCOLS; j += 16) {
        const _Float16* bhi_p = xhi + (size_t)j * UL_D + bstride;
        const _Float16* blo_p = xlo + (size_t)j * UL_D + bstride;

        v8f c8 = {};
#pragma unroll
        for (int c = 0; c < 4; ++c) {
            v16h Bhi = *(const v16h*)(bhi_p + 32 * c);
            v16h Blo = *(const v16h*)(blo_p + 32 * c);
            c8 = __builtin_amdgcn_wmma_f32_16x16x32_f16(
                false, Ahi[c], false, Bhi, (short)0, c8, false, false);
            c8 = __builtin_amdgcn_wmma_f32_16x16x32_f16(
                false, Ahi[c], false, Blo, (short)0, c8, false, false);
            c8 = __builtin_amdgcn_wmma_f32_16x16x32_f16(
                false, Alo[c], false, Bhi, (short)0, c8, false, false);
        }

        // exp(sim/T) via native exp2 (TRANS pipe, co-executes with WMMA).
#pragma unroll
        for (int v = 0; v < 8; ++v)
            acc[v] += __builtin_amdgcn_exp2f(c8[v] * UL_EXP2_SCALE);
    }

    // Reduce across the 16 lanes of each C-matrix half (xor masks < 16 stay
    // within a half). After this every lane of a half holds its row's sum.
#pragma unroll
    for (int v = 0; v < 8; ++v) {
        float s = acc[v];
        s += __shfl_xor(s, 1, 32);
        s += __shfl_xor(s, 2, 32);
        s += __shfl_xor(s, 4, 32);
        s += __shfl_xor(s, 8, 32);
        acc[v] = s;
    }

    // Lane 0 writes rows rowbase+0..7, lane 16 writes rows rowbase+8..15.
    if ((lane & 15) == 0) {
#pragma unroll
        for (int v = 0; v < 8; ++v) {
            int row = rowbase + 8 * kh + v;
            rowpart[(size_t)blockIdx.y * UL_N + row] = acc[v];
        }
    }
}

// ---------------------------------------------------------------------------
// Kernel 3: deterministic reduction: per-row sum of the fixed column slices,
// log, then mean over all rows.
// ---------------------------------------------------------------------------
__global__ void ul_reduce_kernel(const float* __restrict__ rowpart,
                                 float* __restrict__ out) {
    __shared__ float smem[256];
    int tid = threadIdx.x;
    float s = 0.0f;
    for (int i = tid; i < UL_N; i += 256) {
        float t = 0.0f;
#pragma unroll
        for (int p = 0; p < UL_CSPLIT; ++p)
            t += rowpart[(size_t)p * UL_N + i];
        s += __builtin_amdgcn_logf(t) * UL_LN2;
    }
    smem[tid] = s;
    __syncthreads();
    for (int off = 128; off > 0; off >>= 1) {
        if (tid < off) smem[tid] += smem[tid + off];
        __syncthreads();
    }
    if (tid == 0) out[0] = smem[0] * (1.0f / (float)UL_N);
}

extern "C" void kernel_launch(void* const* d_in, const int* in_sizes, int n_in,
                              void* d_out, int out_size, void* d_ws, size_t ws_size,
                              hipStream_t stream) {
    (void)in_sizes; (void)n_in; (void)out_size; (void)ws_size;
    const float* x = (const float*)d_in[0];

    // Workspace layout: xhi (4MB) | xlo (4MB) | rowpart (UL_CSPLIT*64KB)
    _Float16* xhi   = (_Float16*)d_ws;
    _Float16* xlo   = xhi + (size_t)UL_N * UL_D;
    float*    rowpart = (float*)(xlo + (size_t)UL_N * UL_D);

    ul_split_kernel<<<(UL_N * UL_D + 255) / 256, 256, 0, stream>>>(x, xhi, xlo);

    dim3 grid(UL_N / 128, UL_CSPLIT);
    ul_main_kernel<<<grid, 256, 0, stream>>>(xhi, xlo, rowpart);

    ul_reduce_kernel<<<1, 256, 0, stream>>>(rowpart, (float*)d_out);
}
